// SEGNN_20229295964665
// MI455X (gfx1250) — matrix operand order repaired and checked by reading, not compiled, not run on hardware
//
#include <hip/hip_runtime.h>

// ---------------------------------------------------------------------------
// CDNA5 (gfx1250) SEGNN forward.  All dense per-edge / per-node tensor-product
// matmuls run on v_wmma_f32_16x16x32_bf16 (M=16 edges/nodes per wave).
// ---------------------------------------------------------------------------

typedef __bf16 bf16_t;
typedef bf16_t v16bf __attribute__((ext_vector_type(16)));
typedef float  v8f   __attribute__((ext_vector_type(8)));
typedef unsigned int v4u __attribute__((ext_vector_type(4)));

union Frag { v16bf v; v4u q[2]; };

#define EW 4   // waves per block (4 waves = 128 threads, one per SIMD32 of a WGP)

__device__ __forceinline__ unsigned short f2bf(float f) {
  unsigned int u = __float_as_uint(f);
  u += 0x7FFFu + ((u >> 16) & 1u);      // round-to-nearest-even
  return (unsigned short)(u >> 16);
}

// A fragment: 16x32 bf16 tile from LDS (row-major, stride in elements, mult of 8)
// lane<16: row=lane, elems 0..7 = K k0..k0+7, elems 8..15 = K k0+16..k0+23
// lane>=16: row=lane-16, elems 0..7 = K k0+8..15, elems 8..15 = K k0+24..31
__device__ __forceinline__ v16bf ld_afrag(const unsigned short* A, int stride, int k0, int lane) {
  int row = lane & 15, hi = lane >> 4;
  const unsigned short* p = A + row * stride + k0 + hi * 8;
  Frag f;
  f.q[0] = *(const v4u*)(p);
  f.q[1] = *(const v4u*)(p + 16);
  return f.v;
}

// B fragment: 32x16 bf16 tile from transposed weights Wt[N][Kp] in global
// lane<16: col=lane, elems 0..7 = K k0..k0+7, 8..15 = K k0+16..23 ; lane>=16: col=lane-16, K k0+8..15, k0+24..31
__device__ __forceinline__ v16bf ld_bfrag(const unsigned short* Wt, int kp, int k0, int lane) {
  int col = lane & 15, hi = lane >> 4;
  const unsigned short* p = Wt + col * kp + k0 + hi * 8;
  Frag f;
  f.q[0] = *(const v4u*)(p);
  f.q[1] = *(const v4u*)(p + 16);
  return f.v;
}

__device__ __forceinline__ v8f wmma_bf16(v16bf a, v16bf b, v8f c) {
  return __builtin_amdgcn_wmma_f32_16x16x32_bf16(false, a, false, b, (short)0, c, false, false);
}

__device__ __forceinline__ v8f zero8() {
  v8f z;
#pragma unroll
  for (int i = 0; i < 8; ++i) z[i] = 0.f;
  return z;
}

__device__ __forceinline__ float siluf(float x) { return x / (1.f + __expf(-x)); }
__device__ __forceinline__ float sigmf(float x) { return 1.f / (1.f + __expf(-x)); }

// ---------------------------------------------------------------------------
// Weight prep: fp32 [K][N] -> bf16 transposed padded [N][Kp]
// ---------------------------------------------------------------------------
__global__ __launch_bounds__(256) void segnn_prep_kernel(
    const float* __restrict__ src, unsigned short* __restrict__ dst,
    int K, int N, int Kp) {
  int t = blockIdx.x * 256 + threadIdx.x;
  if (t >= N * Kp) return;
  int col = t / Kp, k = t - col * Kp;
  dst[t] = (k < K) ? f2bf(src[(size_t)k * N + col]) : (unsigned short)0;
}

// ---------------------------------------------------------------------------
// Embedding: tiny TP (K=6 scalars, 8 vector paths) -> plain VALU
// ---------------------------------------------------------------------------
__global__ __launch_bounds__(256) void segnn_embed_kernel(
    const float* __restrict__ ns, const float* __restrict__ nv,
    const float* __restrict__ na0, const float* __restrict__ na1,
    const float* __restrict__ W0, const float* __restrict__ W1,
    const float* __restrict__ b0,
    float* __restrict__ S, float* __restrict__ V, int N) {
  int idx = blockIdx.x * 256 + threadIdx.x;
  int n = idx >> 5, o = idx & 31;
  if (n >= N) return;
  float a0 = na0[n];
  float ax = na1[n * 3 + 0], ay = na1[n * 3 + 1], az = na1[n * 3 + 2];
  float s0 = ns[n * 4 + 0], s1 = ns[n * 4 + 1], s2 = ns[n * 4 + 2], s3 = ns[n * 4 + 3];
  float v0x = nv[n * 6 + 0], v0y = nv[n * 6 + 1], v0z = nv[n * 6 + 2];
  float v1x = nv[n * 6 + 3], v1y = nv[n * 6 + 4], v1z = nv[n * 6 + 5];
  float p0[6] = {s0 * a0, s1 * a0, s2 * a0, s3 * a0,
                 v0x * ax + v0y * ay + v0z * az,
                 v1x * ax + v1y * ay + v1z * az};
  float acc = b0[o];
#pragma unroll
  for (int k = 0; k < 6; ++k) acc += p0[k] * W0[k * 32 + o];
  S[(size_t)n * 32 + o] = acc;
  float px[8], py[8], pz[8];
  px[0] = s0 * ax; py[0] = s0 * ay; pz[0] = s0 * az;
  px[1] = s1 * ax; py[1] = s1 * ay; pz[1] = s1 * az;
  px[2] = s2 * ax; py[2] = s2 * ay; pz[2] = s2 * az;
  px[3] = s3 * ax; py[3] = s3 * ay; pz[3] = s3 * az;
  px[4] = v0x * a0; py[4] = v0y * a0; pz[4] = v0z * a0;
  px[5] = v1x * a0; py[5] = v1y * a0; pz[5] = v1z * a0;
  px[6] = v0y * az - v0z * ay; py[6] = v0z * ax - v0x * az; pz[6] = v0x * ay - v0y * ax;
  px[7] = v1y * az - v1z * ay; py[7] = v1z * ax - v1x * az; pz[7] = v1x * ay - v1y * ax;
  float ox = 0.f, oy = 0.f, oz = 0.f;
#pragma unroll
  for (int p = 0; p < 8; ++p) {
    float w = W1[p * 32 + o];
    ox += px[p] * w; oy += py[p] * w; oz += pz[p] * w;
  }
  V[(size_t)n * 96 + o * 3 + 0] = ox;
  V[(size_t)n * 96 + o * 3 + 1] = oy;
  V[(size_t)n * 96 + o * 3 + 2] = oz;
}

// ---------------------------------------------------------------------------
// Edge kernel: per wave 16 edges; msg0 gated TP + msg1 gated TP; atomic scatter
// ---------------------------------------------------------------------------
__global__ __launch_bounds__(128) void segnn_edge_kernel(
    const float* __restrict__ S, const float* __restrict__ V,
    const float* __restrict__ ea0, const float* __restrict__ ea1,
    const float* __restrict__ msgf,
    const int* __restrict__ snd, const int* __restrict__ rcv,
    const unsigned short* __restrict__ m0W0, const unsigned short* __restrict__ m0W1,
    const unsigned short* __restrict__ m1W0, const unsigned short* __restrict__ m1W1,
    const float* __restrict__ m0b, const float* __restrict__ m1b,
    float* __restrict__ aggS, float* __restrict__ aggV, int E) {
  __shared__ unsigned short Atile[EW][16 * 224] __attribute__((aligned(16)));
  __shared__ float SF[EW][16 * 66];
  __shared__ float VF[EW][16 * 192];
  __shared__ float MS[EW][16 * 32];
  __shared__ float MV[EW][16 * 96];
  __shared__ int ESD[EW][16];
  __shared__ int ERC[EW][16];
  __shared__ int EOK[EW][16];
  __shared__ int EIX[EW][16];
  __shared__ float EA0[EW][16];
  __shared__ float EA1[EW][16][3];

  const int tid = threadIdx.x;
  const int w = tid >> 5, lane = tid & 31;
  const int ncol = lane & 15, hi = lane >> 4;
  const int eBase = blockIdx.x * (EW * 16) + w * 16;

  if (lane < 16) {
    int e = eBase + lane;
    bool ok = e < E;
    int ec = ok ? e : 0;
    EIX[w][lane] = ec;
    ESD[w][lane] = ok ? snd[ec] : 0;
    ERC[w][lane] = ok ? rcv[ec] : 0;
    EOK[w][lane] = ok ? 1 : 0;
    EA0[w][lane] = ok ? ea0[ec] : 0.f;
    EA1[w][lane][0] = ok ? ea1[ec * 3 + 0] : 0.f;
    EA1[w][lane][1] = ok ? ea1[ec * 3 + 1] : 0.f;
    EA1[w][lane][2] = ok ? ea1[ec * 3 + 2] : 0.f;
  }
  __syncthreads();

  // gather sfeat (66) and vfeat (64x3)
  for (int t = lane; t < 16 * 66; t += 32) {
    int r = t / 66, k = t - r * 66;
    float val;
    if (k < 32)      val = S[(size_t)ESD[w][r] * 32 + k];
    else if (k < 64) val = S[(size_t)ERC[w][r] * 32 + (k - 32)];
    else             val = msgf[(size_t)EIX[w][r] * 2 + (k - 64)];
    SF[w][t] = val;
  }
  for (int t = lane; t < 16 * 192; t += 32) {
    int r = t / 192, q = t - r * 192;
    int j = q / 3, c = q - j * 3;
    float val = (j < 32) ? V[(size_t)ESD[w][r] * 96 + j * 3 + c]
                         : V[(size_t)ERC[w][r] * 96 + (j - 32) * 3 + c];
    VF[w][t] = val;
  }
  __syncthreads();

  // ---- msg0: p0 (K=130 pad 160) ----
  for (int t = lane; t < 16 * 160; t += 32) {
    int r = t / 160, k = t - r * 160;
    float val = 0.f;
    if (k < 66) val = SF[w][r * 66 + k] * EA0[w][r];
    else if (k < 130) {
      int j = k - 66;
      const float* vv = &VF[w][r * 192 + j * 3];
      val = vv[0] * EA1[w][r][0] + vv[1] * EA1[w][r][1] + vv[2] * EA1[w][r][2];
    }
    Atile[w][r * 224 + k] = f2bf(val);
  }
  __syncthreads();

  v8f sac[4];
#pragma unroll
  for (int i = 0; i < 4; ++i) sac[i] = zero8();
#pragma unroll
  for (int kt = 0; kt < 5; ++kt) {
    v16bf a = ld_afrag(&Atile[w][0], 224, kt * 32, lane);
#pragma unroll
    for (int nt = 0; nt < 4; ++nt) {
      v16bf b = ld_bfrag(m0W0 + nt * 16 * 160, 160, kt * 32, lane);
      sac[nt] = wmma_bf16(a, b, sac[nt]);
    }
  }
  v8f gate0[2];
#pragma unroll
  for (int nt = 0; nt < 2; ++nt) {
    float bs = m0b[nt * 16 + ncol];
    float bg = m0b[32 + nt * 16 + ncol];
#pragma unroll
    for (int r = 0; r < 8; ++r) {
      int m = r + hi * 8;
      MS[w][m * 32 + nt * 16 + ncol] = siluf(sac[nt][r] + bs);
      gate0[nt][r] = sigmf(sac[2 + nt][r] + bg);
    }
  }
  __syncthreads();

  // ---- msg0: p1 per component (K=194 pad 224) ----
  for (int c = 0; c < 3; ++c) {
    int c1 = (c + 1) % 3, c2 = (c + 2) % 3;
    for (int t = lane; t < 16 * 224; t += 32) {
      int r = t / 224, k = t - r * 224;
      float val = 0.f;
      if (k < 66) val = SF[w][r * 66 + k] * EA1[w][r][c];
      else if (k < 130) val = VF[w][r * 192 + (k - 66) * 3 + c] * EA0[w][r];
      else if (k < 194) {
        int j = k - 130;
        val = VF[w][r * 192 + j * 3 + c1] * EA1[w][r][c2]
            - VF[w][r * 192 + j * 3 + c2] * EA1[w][r][c1];
      }
      Atile[w][r * 224 + k] = f2bf(val);
    }
    __syncthreads();
    v8f vac[2]; vac[0] = zero8(); vac[1] = zero8();
#pragma unroll
    for (int kt = 0; kt < 7; ++kt) {
      v16bf a = ld_afrag(&Atile[w][0], 224, kt * 32, lane);
#pragma unroll
      for (int nt = 0; nt < 2; ++nt) {
        v16bf b = ld_bfrag(m0W1 + nt * 16 * 224, 224, kt * 32, lane);
        vac[nt] = wmma_bf16(a, b, vac[nt]);
      }
    }
#pragma unroll
    for (int nt = 0; nt < 2; ++nt)
#pragma unroll
      for (int r = 0; r < 8; ++r) {
        int m = r + hi * 8;
        MV[w][m * 96 + (nt * 16 + ncol) * 3 + c] = vac[nt][r] * gate0[nt][r];
      }
    __syncthreads();
  }

  // ---- msg1: p0 (K=64) ----
  for (int t = lane; t < 16 * 64; t += 32) {
    int r = t >> 6, k = t & 63;
    float val;
    if (k < 32) val = MS[w][r * 32 + k] * EA0[w][r];
    else {
      int j = k - 32;
      const float* vv = &MV[w][r * 96 + j * 3];
      val = vv[0] * EA1[w][r][0] + vv[1] * EA1[w][r][1] + vv[2] * EA1[w][r][2];
    }
    Atile[w][r * 64 + k] = f2bf(val);
  }
  __syncthreads();
#pragma unroll
  for (int i = 0; i < 4; ++i) sac[i] = zero8();
#pragma unroll
  for (int kt = 0; kt < 2; ++kt) {
    v16bf a = ld_afrag(&Atile[w][0], 64, kt * 32, lane);
#pragma unroll
    for (int nt = 0; nt < 4; ++nt) {
      v16bf b = ld_bfrag(m1W0 + nt * 16 * 64, 64, kt * 32, lane);
      sac[nt] = wmma_bf16(a, b, sac[nt]);
    }
  }
  v8f gate1[2];
#pragma unroll
  for (int nt = 0; nt < 2; ++nt) {
    float bs = m1b[nt * 16 + ncol];
    float bg = m1b[32 + nt * 16 + ncol];
#pragma unroll
    for (int r = 0; r < 8; ++r) {
      int m = r + hi * 8;
      float sv = siluf(sac[nt][r] + bs);
      if (EOK[w][m]) atomicAdd(&aggS[(size_t)ERC[w][m] * 32 + nt * 16 + ncol], sv);
      gate1[nt][r] = sigmf(sac[2 + nt][r] + bg);
    }
  }
  __syncthreads();

  // ---- msg1: p1 per component (K=96) ----
  for (int c = 0; c < 3; ++c) {
    int c1 = (c + 1) % 3, c2 = (c + 2) % 3;
    for (int t = lane; t < 16 * 96; t += 32) {
      int r = t / 96, k = t - r * 96;
      float val;
      if (k < 32) val = MS[w][r * 32 + k] * EA1[w][r][c];
      else if (k < 64) val = MV[w][r * 96 + (k - 32) * 3 + c] * EA0[w][r];
      else {
        int j = k - 64;
        val = MV[w][r * 96 + j * 3 + c1] * EA1[w][r][c2]
            - MV[w][r * 96 + j * 3 + c2] * EA1[w][r][c1];
      }
      Atile[w][r * 96 + k] = f2bf(val);
    }
    __syncthreads();
    v8f vac[2]; vac[0] = zero8(); vac[1] = zero8();
#pragma unroll
    for (int kt = 0; kt < 3; ++kt) {
      v16bf a = ld_afrag(&Atile[w][0], 96, kt * 32, lane);
#pragma unroll
      for (int nt = 0; nt < 2; ++nt) {
        v16bf b = ld_bfrag(m1W1 + nt * 16 * 96, 96, kt * 32, lane);
        vac[nt] = wmma_bf16(a, b, vac[nt]);
      }
    }
#pragma unroll
    for (int nt = 0; nt < 2; ++nt)
#pragma unroll
      for (int r = 0; r < 8; ++r) {
        int m = r + hi * 8;
        if (EOK[w][m])
          atomicAdd(&aggV[(size_t)ERC[w][m] * 96 + (nt * 16 + ncol) * 3 + c],
                    vac[nt][r] * gate1[nt][r]);
      }
    __syncthreads();
  }
}

// ---------------------------------------------------------------------------
// Node update: upd0 gated TP + upd1 TP + residual, 16 nodes per wave
// ---------------------------------------------------------------------------
__global__ __launch_bounds__(128) void segnn_node_kernel(
    float* __restrict__ S, float* __restrict__ V,
    const float* __restrict__ aggS, const float* __restrict__ aggV,
    const float* __restrict__ na0, const float* __restrict__ na1,
    const unsigned short* __restrict__ u0W0, const unsigned short* __restrict__ u0W1,
    const unsigned short* __restrict__ u1W0, const unsigned short* __restrict__ u1W1,
    const float* __restrict__ u0b, const float* __restrict__ u1b, int N) {
  __shared__ unsigned short Atile[EW][16 * 192] __attribute__((aligned(16)));
  __shared__ float USf[EW][16 * 64];
  __shared__ float UVf[EW][16 * 192];
  __shared__ float US2[EW][16 * 32];
  __shared__ float UV2[EW][16 * 96];
  __shared__ int NID[EW][16];
  __shared__ float A0s[EW][16];
  __shared__ float A1s[EW][16][3];

  const int tid = threadIdx.x;
  const int w = tid >> 5, lane = tid & 31;
  const int ncol = lane & 15, hi = lane >> 4;
  const int nBase = blockIdx.x * (EW * 16) + w * 16;

  if (lane < 16) {
    int n = nBase + lane;
    bool ok = n < N;
    int nc = ok ? n : 0;
    NID[w][lane] = ok ? n : -1;
    A0s[w][lane] = ok ? na0[nc] : 0.f;
    A1s[w][lane][0] = ok ? na1[nc * 3 + 0] : 0.f;
    A1s[w][lane][1] = ok ? na1[nc * 3 + 1] : 0.f;
    A1s[w][lane][2] = ok ? na1[nc * 3 + 2] : 0.f;
  }
  __syncthreads();
  for (int t = lane; t < 16 * 64; t += 32) {
    int r = t >> 6, k = t & 63;
    int n = NID[w][r];
    float val = 0.f;
    if (n >= 0) val = (k < 32) ? S[(size_t)n * 32 + k] : aggS[(size_t)n * 32 + (k - 32)];
    USf[w][t] = val;
  }
  for (int t = lane; t < 16 * 192; t += 32) {
    int r = t / 192, q = t - r * 192;
    int n = NID[w][r];
    float val = 0.f;
    if (n >= 0) {
      int j = q / 3, c = q - j * 3;
      val = (j < 32) ? V[(size_t)n * 96 + j * 3 + c] : aggV[(size_t)n * 96 + (j - 32) * 3 + c];
    }
    UVf[w][t] = val;
  }
  __syncthreads();

  // upd0 p0 (K=128)
  for (int t = lane; t < 16 * 128; t += 32) {
    int r = t >> 7, k = t & 127;
    float val;
    if (k < 64) val = USf[w][r * 64 + k] * A0s[w][r];
    else {
      int j = k - 64;
      const float* vv = &UVf[w][r * 192 + j * 3];
      val = vv[0] * A1s[w][r][0] + vv[1] * A1s[w][r][1] + vv[2] * A1s[w][r][2];
    }
    Atile[w][r * 128 + k] = f2bf(val);
  }
  __syncthreads();
  v8f sac[4];
#pragma unroll
  for (int i = 0; i < 4; ++i) sac[i] = zero8();
#pragma unroll
  for (int kt = 0; kt < 4; ++kt) {
    v16bf a = ld_afrag(&Atile[w][0], 128, kt * 32, lane);
#pragma unroll
    for (int nt = 0; nt < 4; ++nt) {
      v16bf b = ld_bfrag(u0W0 + nt * 16 * 128, 128, kt * 32, lane);
      sac[nt] = wmma_bf16(a, b, sac[nt]);
    }
  }
  v8f gate[2];
#pragma unroll
  for (int nt = 0; nt < 2; ++nt) {
    float bs = u0b[nt * 16 + ncol];
    float bg = u0b[32 + nt * 16 + ncol];
#pragma unroll
    for (int r = 0; r < 8; ++r) {
      int m = r + hi * 8;
      US2[w][m * 32 + nt * 16 + ncol] = siluf(sac[nt][r] + bs);
      gate[nt][r] = sigmf(sac[2 + nt][r] + bg);
    }
  }
  __syncthreads();

  // upd0 p1 per component (K=192)
  for (int c = 0; c < 3; ++c) {
    int c1 = (c + 1) % 3, c2 = (c + 2) % 3;
    for (int t = lane; t < 16 * 192; t += 32) {
      int r = t / 192, k = t - r * 192;
      float val;
      if (k < 64) val = USf[w][r * 64 + k] * A1s[w][r][c];
      else if (k < 128) val = UVf[w][r * 192 + (k - 64) * 3 + c] * A0s[w][r];
      else {
        int j = k - 128;
        val = UVf[w][r * 192 + j * 3 + c1] * A1s[w][r][c2]
            - UVf[w][r * 192 + j * 3 + c2] * A1s[w][r][c1];
      }
      Atile[w][r * 192 + k] = f2bf(val);
    }
    __syncthreads();
    v8f vac[2]; vac[0] = zero8(); vac[1] = zero8();
#pragma unroll
    for (int kt = 0; kt < 6; ++kt) {
      v16bf a = ld_afrag(&Atile[w][0], 192, kt * 32, lane);
#pragma unroll
      for (int nt = 0; nt < 2; ++nt) {
        v16bf b = ld_bfrag(u0W1 + nt * 16 * 192, 192, kt * 32, lane);
        vac[nt] = wmma_bf16(a, b, vac[nt]);
      }
    }
#pragma unroll
    for (int nt = 0; nt < 2; ++nt)
#pragma unroll
      for (int r = 0; r < 8; ++r) {
        int m = r + hi * 8;
        UV2[w][m * 96 + (nt * 16 + ncol) * 3 + c] = vac[nt][r] * gate[nt][r];
      }
    __syncthreads();
  }

  // upd1 p0 (K=64) -> residual into S
  for (int t = lane; t < 16 * 64; t += 32) {
    int r = t >> 6, k = t & 63;
    float val;
    if (k < 32) val = US2[w][r * 32 + k] * A0s[w][r];
    else {
      int j = k - 32;
      const float* vv = &UV2[w][r * 96 + j * 3];
      val = vv[0] * A1s[w][r][0] + vv[1] * A1s[w][r][1] + vv[2] * A1s[w][r][2];
    }
    Atile[w][r * 64 + k] = f2bf(val);
  }
  __syncthreads();
  v8f so[2]; so[0] = zero8(); so[1] = zero8();
#pragma unroll
  for (int kt = 0; kt < 2; ++kt) {
    v16bf a = ld_afrag(&Atile[w][0], 64, kt * 32, lane);
#pragma unroll
    for (int nt = 0; nt < 2; ++nt) {
      v16bf b = ld_bfrag(u1W0 + nt * 16 * 64, 64, kt * 32, lane);
      so[nt] = wmma_bf16(a, b, so[nt]);
    }
  }
#pragma unroll
  for (int nt = 0; nt < 2; ++nt) {
    float bb = u1b[nt * 16 + ncol];
#pragma unroll
    for (int r = 0; r < 8; ++r) {
      int m = r + hi * 8;
      int n = NID[w][m];
      if (n >= 0) {
        int col = nt * 16 + ncol;
        S[(size_t)n * 32 + col] = USf[w][m * 64 + col] + so[nt][r] + bb;
      }
    }
  }
  __syncthreads();

  // upd1 p1 per component (K=96) -> residual into V
  for (int c = 0; c < 3; ++c) {
    int c1 = (c + 1) % 3, c2 = (c + 2) % 3;
    for (int t = lane; t < 16 * 96; t += 32) {
      int r = t / 96, k = t - r * 96;
      float val;
      if (k < 32) val = US2[w][r * 32 + k] * A1s[w][r][c];
      else if (k < 64) val = UV2[w][r * 96 + (k - 32) * 3 + c] * A0s[w][r];
      else {
        int j = k - 64;
        val = UV2[w][r * 96 + j * 3 + c1] * A1s[w][r][c2]
            - UV2[w][r * 96 + j * 3 + c2] * A1s[w][r][c1];
      }
      Atile[w][r * 96 + k] = f2bf(val);
    }
    __syncthreads();
    v8f vo[2]; vo[0] = zero8(); vo[1] = zero8();
#pragma unroll
    for (int kt = 0; kt < 3; ++kt) {
      v16bf a = ld_afrag(&Atile[w][0], 96, kt * 32, lane);
#pragma unroll
      for (int nt = 0; nt < 2; ++nt) {
        v16bf b = ld_bfrag(u1W1 + nt * 16 * 96, 96, kt * 32, lane);
        vo[nt] = wmma_bf16(a, b, vo[nt]);
      }
    }
#pragma unroll
    for (int nt = 0; nt < 2; ++nt)
#pragma unroll
      for (int r = 0; r < 8; ++r) {
        int m = r + hi * 8;
        int n = NID[w][m];
        if (n >= 0) {
          int jj = nt * 16 + ncol;
          V[(size_t)n * 96 + jj * 3 + c] = UVf[w][m * 192 + jj * 3 + c] + vo[nt][r];
        }
      }
    __syncthreads();
  }
}

// ---------------------------------------------------------------------------
// Prepool: prepool0 gated TP + prepool1 TP-to-scalars, atomics into graph sums
// ---------------------------------------------------------------------------
__global__ __launch_bounds__(128) void segnn_prepool_kernel(
    const float* __restrict__ S, const float* __restrict__ V,
    const float* __restrict__ na0, const float* __restrict__ na1,
    const int* __restrict__ gid,
    const unsigned short* __restrict__ q0W0, const unsigned short* __restrict__ q0W1,
    const unsigned short* __restrict__ q1W0,
    const float* __restrict__ q0b, const float* __restrict__ q1b,
    float* __restrict__ gsum, float* __restrict__ gcnt, int N) {
  __shared__ unsigned short Atile[EW][16 * 96] __attribute__((aligned(16)));
  __shared__ float SFp[EW][16 * 32];
  __shared__ float VFp[EW][16 * 96];
  __shared__ float S2[EW][16 * 32];
  __shared__ float V2[EW][16 * 96];
  __shared__ int NID[EW][16];
  __shared__ int GIDs[EW][16];
  __shared__ float A0s[EW][16];
  __shared__ float A1s[EW][16][3];

  const int tid = threadIdx.x;
  const int w = tid >> 5, lane = tid & 31;
  const int ncol = lane & 15, hi = lane >> 4;
  const int nBase = blockIdx.x * (EW * 16) + w * 16;

  if (lane < 16) {
    int n = nBase + lane;
    bool ok = n < N;
    int nc = ok ? n : 0;
    NID[w][lane] = ok ? n : -1;
    int g = ok ? gid[nc] : 0;
    GIDs[w][lane] = g;
    A0s[w][lane] = ok ? na0[nc] : 0.f;
    A1s[w][lane][0] = ok ? na1[nc * 3 + 0] : 0.f;
    A1s[w][lane][1] = ok ? na1[nc * 3 + 1] : 0.f;
    A1s[w][lane][2] = ok ? na1[nc * 3 + 2] : 0.f;
    if (ok) atomicAdd(&gcnt[g], 1.f);
  }
  __syncthreads();
  for (int t = lane; t < 16 * 32; t += 32) {
    int r = t >> 5, k = t & 31;
    int n = NID[w][r];
    SFp[w][t] = (n >= 0) ? S[(size_t)n * 32 + k] : 0.f;
  }
  for (int t = lane; t < 16 * 96; t += 32) {
    int r = t / 96, q = t - r * 96;
    int n = NID[w][r];
    VFp[w][t] = (n >= 0) ? V[(size_t)n * 96 + q] : 0.f;
  }
  __syncthreads();

  // prepool0 p0 (K=64)
  for (int t = lane; t < 16 * 64; t += 32) {
    int r = t >> 6, k = t & 63;
    float val;
    if (k < 32) val = SFp[w][r * 32 + k] * A0s[w][r];
    else {
      int j = k - 32;
      const float* vv = &VFp[w][r * 96 + j * 3];
      val = vv[0] * A1s[w][r][0] + vv[1] * A1s[w][r][1] + vv[2] * A1s[w][r][2];
    }
    Atile[w][r * 64 + k] = f2bf(val);
  }
  __syncthreads();
  v8f sac[4];
#pragma unroll
  for (int i = 0; i < 4; ++i) sac[i] = zero8();
#pragma unroll
  for (int kt = 0; kt < 2; ++kt) {
    v16bf a = ld_afrag(&Atile[w][0], 64, kt * 32, lane);
#pragma unroll
    for (int nt = 0; nt < 4; ++nt) {
      v16bf b = ld_bfrag(q0W0 + nt * 16 * 64, 64, kt * 32, lane);
      sac[nt] = wmma_bf16(a, b, sac[nt]);
    }
  }
  v8f gate[2];
#pragma unroll
  for (int nt = 0; nt < 2; ++nt) {
    float bs = q0b[nt * 16 + ncol];
    float bg = q0b[32 + nt * 16 + ncol];
#pragma unroll
    for (int r = 0; r < 8; ++r) {
      int m = r + hi * 8;
      S2[w][m * 32 + nt * 16 + ncol] = siluf(sac[nt][r] + bs);
      gate[nt][r] = sigmf(sac[2 + nt][r] + bg);
    }
  }
  __syncthreads();

  // prepool0 p1 per component (K=96)
  for (int c = 0; c < 3; ++c) {
    int c1 = (c + 1) % 3, c2 = (c + 2) % 3;
    for (int t = lane; t < 16 * 96; t += 32) {
      int r = t / 96, k = t - r * 96;
      float val;
      if (k < 32) val = SFp[w][r * 32 + k] * A1s[w][r][c];
      else if (k < 64) val = VFp[w][r * 96 + (k - 32) * 3 + c] * A0s[w][r];
      else {
        int j = k - 64;
        val = VFp[w][r * 96 + j * 3 + c1] * A1s[w][r][c2]
            - VFp[w][r * 96 + j * 3 + c2] * A1s[w][r][c1];
      }
      Atile[w][r * 96 + k] = f2bf(val);
    }
    __syncthreads();
    v8f vac[2]; vac[0] = zero8(); vac[1] = zero8();
#pragma unroll
    for (int kt = 0; kt < 3; ++kt) {
      v16bf a = ld_afrag(&Atile[w][0], 96, kt * 32, lane);
#pragma unroll
      for (int nt = 0; nt < 2; ++nt) {
        v16bf b = ld_bfrag(q0W1 + nt * 16 * 96, 96, kt * 32, lane);
        vac[nt] = wmma_bf16(a, b, vac[nt]);
      }
    }
#pragma unroll
    for (int nt = 0; nt < 2; ++nt)
#pragma unroll
      for (int r = 0; r < 8; ++r) {
        int m = r + hi * 8;
        V2[w][m * 96 + (nt * 16 + ncol) * 3 + c] = vac[nt][r] * gate[nt][r];
      }
    __syncthreads();
  }

  // prepool1 p0 (K=64) -> scalars, atomic into graph sums
  for (int t = lane; t < 16 * 64; t += 32) {
    int r = t >> 6, k = t & 63;
    float val;
    if (k < 32) val = S2[w][r * 32 + k] * A0s[w][r];
    else {
      int j = k - 32;
      const float* vv = &V2[w][r * 96 + j * 3];
      val = vv[0] * A1s[w][r][0] + vv[1] * A1s[w][r][1] + vv[2] * A1s[w][r][2];
    }
    Atile[w][r * 64 + k] = f2bf(val);
  }
  __syncthreads();
  v8f pac[4];
#pragma unroll
  for (int i = 0; i < 4; ++i) pac[i] = zero8();
#pragma unroll
  for (int kt = 0; kt < 2; ++kt) {
    v16bf a = ld_afrag(&Atile[w][0], 64, kt * 32, lane);
#pragma unroll
    for (int nt = 0; nt < 4; ++nt) {
      v16bf b = ld_bfrag(q1W0 + nt * 16 * 64, 64, kt * 32, lane);
      pac[nt] = wmma_bf16(a, b, pac[nt]);
    }
  }
#pragma unroll
  for (int nt = 0; nt < 4; ++nt) {
    float bb = q1b[nt * 16 + ncol];
#pragma unroll
    for (int r = 0; r < 8; ++r) {
      int m = r + hi * 8;
      if (NID[w][m] >= 0)
        atomicAdd(&gsum[(size_t)GIDs[w][m] * 64 + nt * 16 + ncol], pac[nt][r] + bb);
    }
  }
}

// ---------------------------------------------------------------------------
// Final: graph mean -> silu(g@postW+b) -> @outW+b
// ---------------------------------------------------------------------------
__global__ __launch_bounds__(256) void segnn_final_kernel(
    const float* __restrict__ gsum, const float* __restrict__ gcnt,
    const float* __restrict__ postW, const float* __restrict__ postB,
    const float* __restrict__ outW, const float* __restrict__ outB,
    float* __restrict__ out) {
  __shared__ float H[100 * 64];
  int tid = threadIdx.x;
  for (int idx = tid; idx < 100 * 64; idx += 256) {
    int g = idx >> 6, o = idx & 63;
    float cnt = fmaxf(gcnt[g], 1.f);
    float acc = postB[o];
    for (int k = 0; k < 64; ++k) acc += (gsum[g * 64 + k] / cnt) * postW[k * 64 + o];
    H[idx] = acc / (1.f + __expf(-acc));
  }
  __syncthreads();
  for (int g = tid; g < 100; g += 256) {
    float acc = outB[0];
    for (int o = 0; o < 64; ++o) acc += H[g * 64 + o] * outW[o];
    out[g] = acc;
  }
}

// ---------------------------------------------------------------------------
// Host launcher
// ---------------------------------------------------------------------------
extern "C" void kernel_launch(void* const* d_in, const int* in_sizes, int n_in,
                              void* d_out, int out_size, void* d_ws, size_t ws_size,
                              hipStream_t stream) {
  (void)n_in; (void)out_size; (void)ws_size;
  const int N = in_sizes[2];   // node_attr_s count == N_NODES
  const int E = in_sizes[4];   // edge_attr_s count == N_EDGES

  const float* node_s = (const float*)d_in[0];
  const float* node_v = (const float*)d_in[1];
  const float* na0    = (const float*)d_in[2];
  const float* na1    = (const float*)d_in[3];
  const float* ea0    = (const float*)d_in[4];
  const float* ea1    = (const float*)d_in[5];
  const float* msgf   = (const float*)d_in[6];
  const int*   snd    = (const int*)d_in[7];
  const int*   rcv    = (const int*)d_in[8];
  const int*   gid    = (const int*)d_in[9];
  // params flattened with sorted dict keys (jax pytree order):
  // embed{W0,W1,b0}=10..12, layers[l]{msg0,msg1,upd0,upd1}{W0,W1,b0}=13+12l..,
  // output{W,b}=61,62, postpool0{W,b}=63,64, prepool0{W0,W1,b0}=65..67,
  // prepool1{W0,b0}=68,69
  const float* eW0   = (const float*)d_in[10];
  const float* eW1   = (const float*)d_in[11];
  const float* eb0   = (const float*)d_in[12];
  const float* outW  = (const float*)d_in[61];
  const float* outB  = (const float*)d_in[62];
  const float* postW = (const float*)d_in[63];
  const float* postB = (const float*)d_in[64];
  const float* pp0W0 = (const float*)d_in[65];
  const float* pp0W1 = (const float*)d_in[66];
  const float* pp0b  = (const float*)d_in[67];
  const float* pp1W0 = (const float*)d_in[68];
  const float* pp1b  = (const float*)d_in[69];

  char* ws = (char*)d_ws;
  size_t off = 0;
  auto take = [&](size_t bytes) -> char* {
    char* p = ws + off;
    off = (off + bytes + 255) & ~(size_t)255;
    return p;
  };
  float* S    = (float*)take((size_t)N * 32 * 4);
  float* Vv   = (float*)take((size_t)N * 96 * 4);
  float* aggS = (float*)take((size_t)N * 32 * 4);
  float* aggV = (float*)take((size_t)N * 96 * 4);
  float* gsum = (float*)take(100 * 64 * 4);
  float* gcnt = (float*)take(100 * 4);
  unsigned short* WB = (unsigned short*)take((size_t)187392 * 2);

  auto prep = [&](const float* src, unsigned short* dst, int K, int Nc, int Kp) {
    int total = Nc * Kp;
    segnn_prep_kernel<<<(total + 255) / 256, 256, 0, stream>>>(src, dst, K, Nc, Kp);
  };

  // bf16 transposed weight block offsets (elements)
  const int OFF_M0W0 = 0,       OFF_M0W1 = 10240, OFF_M1W0 = 17408, OFF_M1W1 = 21504,
            OFF_U0W0 = 24576,   OFF_U0W1 = 32768, OFF_U1W0 = 38912, OFF_U1W1 = 40960,
            LAYER_W  = 44032;
  for (int l = 0; l < 4; ++l) {
    int b = 13 + 12 * l;
    unsigned short* LW = WB + (size_t)l * LAYER_W;
    prep((const float*)d_in[b + 0],  LW + OFF_M0W0, 130, 64, 160);
    prep((const float*)d_in[b + 1],  LW + OFF_M0W1, 194, 32, 224);
    prep((const float*)d_in[b + 3],  LW + OFF_M1W0,  64, 64,  64);
    prep((const float*)d_in[b + 4],  LW + OFF_M1W1,  96, 32,  96);
    prep((const float*)d_in[b + 6],  LW + OFF_U0W0, 128, 64, 128);
    prep((const float*)d_in[b + 7],  LW + OFF_U0W1, 192, 32, 192);
    prep((const float*)d_in[b + 9],  LW + OFF_U1W0,  64, 32,  64);
    prep((const float*)d_in[b + 10], LW + OFF_U1W1,  96, 32,  96);
  }
  unsigned short* P0W0 = WB + (size_t)4 * LAYER_W;
  unsigned short* P0W1 = P0W0 + 4096;
  unsigned short* P1W0 = P0W1 + 3072;
  prep(pp0W0, P0W0, 64, 64, 64);
  prep(pp0W1, P0W1, 96, 32, 96);
  prep(pp1W0, P1W0, 64, 64, 64);

  // embedding
  {
    int total = N * 32;
    segnn_embed_kernel<<<(total + 255) / 256, 256, 0, stream>>>(
        node_s, node_v, na0, na1, eW0, eW1, eb0, S, Vv, N);
  }

  int eblocks = (E + EW * 16 - 1) / (EW * 16);
  int nblocks = (N + EW * 16 - 1) / (EW * 16);
  for (int l = 0; l < 4; ++l) {
    int b = 13 + 12 * l;
    unsigned short* LW = WB + (size_t)l * LAYER_W;
    hipMemsetAsync(aggS, 0, (size_t)N * 32 * 4, stream);
    hipMemsetAsync(aggV, 0, (size_t)N * 96 * 4, stream);
    segnn_edge_kernel<<<eblocks, 128, 0, stream>>>(
        S, Vv, ea0, ea1, msgf, snd, rcv,
        LW + OFF_M0W0, LW + OFF_M0W1, LW + OFF_M1W0, LW + OFF_M1W1,
        (const float*)d_in[b + 2], (const float*)d_in[b + 5], aggS, aggV, E);
    segnn_node_kernel<<<nblocks, 128, 0, stream>>>(
        S, Vv, aggS, aggV, na0, na1,
        LW + OFF_U0W0, LW + OFF_U0W1, LW + OFF_U1W0, LW + OFF_U1W1,
        (const float*)d_in[b + 8], (const float*)d_in[b + 11], N);
  }

  hipMemsetAsync(gsum, 0, 100 * 64 * 4, stream);
  hipMemsetAsync(gcnt, 0, 100 * 4, stream);
  segnn_prepool_kernel<<<nblocks, 128, 0, stream>>>(
      S, Vv, na0, na1, gid, P0W0, P0W1, P1W0, pp0b, pp1b, gsum, gcnt, N);
  segnn_final_kernel<<<1, 256, 0, stream>>>(
      gsum, gcnt, postW, postB, outW, outB, (float*)d_out);
}
